// LSTMDSTGCRN_78958678770248
// MI455X (gfx1250) — matrix operand order repaired and checked by reading, not compiled, not run on hardware
//
#include <hip/hip_runtime.h>
#include <hip/hip_bf16.h>

typedef __attribute__((ext_vector_type(16))) _Float16 v16h;
typedef __attribute__((ext_vector_type(8)))  _Float16 v8h;
typedef __attribute__((ext_vector_type(8)))  float    v8f;

#define NP 1024     // padded node count
#define NN 1000     // nodes
#define BB 64       // batch
#define HH 64       // hidden
#define TT 48       // time steps
#define GG 256      // 4H gates
#define CI 2        // input channels
#define HOR 12      // horizon

// ---- WMMA fragment loaders (contiguous k = 16h+s, same in A and B) --------------------
// A-operand 16x32 f16: lanes 0-15 (M=lane): e0..15=K0..15
//                      lanes16-31 (M=lane-16): e0..15=K16..31
__device__ __forceinline__ v16h load_afrag(const _Float16* __restrict__ base,
                                           int ld, int row, int k0, int lane) {
  const _Float16* p = base + (size_t)row * ld + k0 + ((lane >> 4) & 1) * 16;
  v8h lo = *(const v8h*)(p);
  v8h hi = *(const v8h*)(p + 8);
  v16h a;
#pragma unroll
  for (int e = 0; e < 8; e++) { a[e] = lo[e]; a[e + 8] = hi[e]; }
  return a;
}

// B-operand 32x16 f16 from TRANSPOSED storage Bt[N][K]: lane n=lane&15,
// lanes<16: K=k0..k0+15 ; lanes>=16: K=k0+16..k0+31  -> 16 contiguous halves.
__device__ __forceinline__ v16h load_bfrag(const _Float16* __restrict__ baseT,
                                           int ld, int col, int k0, int lane) {
  const _Float16* p = baseT + (size_t)col * ld + k0 + ((lane >> 4) & 1) * 16;
  return *(const v16h*)(p);
}

// ---- A = softmax(relu(E1 E2^T)) row-wise, zero-padded to 1024x1024 f16 -----
__global__ void build_A(const float* __restrict__ E1, const float* __restrict__ E2,
                        _Float16* __restrict__ Af) {
  __shared__ float red[256];
  const int i = blockIdx.x;
  const int tid = threadIdx.x;
  if (i >= NN) {
    for (int j = tid; j < NP; j += 256) Af[(size_t)i * NP + j] = (_Float16)0.f;
    return;
  }
  float e1[16];
#pragma unroll
  for (int e = 0; e < 16; e++) e1[e] = E1[i * 16 + e];
  float v[4];
  float mx = 0.f;
#pragma unroll
  for (int q = 0; q < 4; q++) {
    int j = tid + q * 256;
    float s = -1e30f;
    if (j < NN) {
      s = 0.f;
#pragma unroll
      for (int e = 0; e < 16; e++) s += e1[e] * E2[j * 16 + e];
      s = fmaxf(s, 0.f);
      mx = fmaxf(mx, s);
    }
    v[q] = s;
  }
  red[tid] = mx; __syncthreads();
  for (int st = 128; st > 0; st >>= 1) {
    if (tid < st) red[tid] = fmaxf(red[tid], red[tid + st]);
    __syncthreads();
  }
  const float gmax = red[0]; __syncthreads();
  float sum = 0.f;
#pragma unroll
  for (int q = 0; q < 4; q++) {
    int j = tid + q * 256;
    float e = (j < NN) ? __expf(v[q] - gmax) : 0.f;
    v[q] = e; sum += e;
  }
  red[tid] = sum; __syncthreads();
  for (int st = 128; st > 0; st >>= 1) {
    if (tid < st) red[tid] += red[tid + st];
    __syncthreads();
  }
  const float inv = 1.f / red[0];
#pragma unroll
  for (int q = 0; q < 4; q++) {
    int j = tid + q * 256;
    Af[(size_t)i * NP + j] = (_Float16)((j < NN) ? v[q] * inv : 0.f);
  }
}

// ---- Xs[t][b*2+c][j] = x[b,t,j,c] transposed+f16, zero-padded --------------
__global__ void build_Xs(const float* __restrict__ x, _Float16* __restrict__ Xs) {
  const int idx = blockIdx.x * 256 + threadIdx.x;   // t*128*1024 total
  const int j  = idx & (NP - 1);
  const int cc = (idx >> 10) & 127;
  const int t  = idx >> 17;
  if (t >= TT) return;
  const int b = cc >> 1, c = cc & 1;
  float v = 0.f;
  if (j < NN) v = x[(((size_t)b * TT + t) * NN + j) * CI + c];
  Xs[idx] = (_Float16)v;
}

__global__ void f32_to_f16(const float* __restrict__ s, _Float16* __restrict__ d, int n) {
  int i = blockIdx.x * 256 + threadIdx.x;
  if (i < n) d[i] = (_Float16)s[i];
}

// ---- Big diffusion GEMM: Out[M=1024][Ncols] = Af[1024][1024] @ Bt^T --------
// Bt stored transposed [Ncols][1024]. 8 waves/block; each wave register-blocks
// 2 row-tiles x 4 col-tiles (32x64 output): 12 b128 loads per 8 WMMAs.
template <bool F16OUT>
__global__ void wmma_gemm_k1024(const _Float16* __restrict__ A,
                                const _Float16* __restrict__ Bt,
                                void* __restrict__ Out, int Ncols) {
  const int lane = threadIdx.x & 31;
  const int wave = threadIdx.x >> 5;
  const int m0   = blockIdx.x * 256 + wave * 32;   // two 16-row tiles
  const int n0   = blockIdx.y * 64;                // four 16-col tiles
  const int r0   = m0 + (lane & 15);

  v8f acc[2][4];
#pragma unroll
  for (int s = 0; s < 2; s++)
#pragma unroll
    for (int t = 0; t < 4; t++)
#pragma unroll
      for (int e = 0; e < 8; e++) acc[s][t][e] = 0.f;

  for (int k0 = 0; k0 < NP; k0 += 32) {
    v16h a0 = load_afrag(A, NP, r0,      k0, lane);
    v16h a1 = load_afrag(A, NP, r0 + 16, k0, lane);
#pragma unroll
    for (int t = 0; t < 4; t++) {
      v16h b = load_bfrag(Bt, NP, n0 + t * 16 + (lane & 15), k0, lane);
      acc[0][t] = __builtin_amdgcn_wmma_f32_16x16x32_f16(
          false, a0, false, b, (short)0, acc[0][t], false, false);
      acc[1][t] = __builtin_amdgcn_wmma_f32_16x16x32_f16(
          false, a1, false, b, (short)0, acc[1][t], false, false);
    }
  }
#pragma unroll
  for (int s = 0; s < 2; s++) {
    const int mbase = m0 + s * 16 + ((lane >> 4) & 1) * 8;
#pragma unroll
    for (int t = 0; t < 4; t++) {
      const int n = n0 + t * 16 + (lane & 15);
#pragma unroll
      for (int r = 0; r < 8; r++) {
        const size_t o = (size_t)(mbase + r) * Ncols + n;
        if constexpr (F16OUT) ((_Float16*)Out)[o] = (_Float16)acc[s][t][r];
        else                  ((float*)Out)[o]    = acc[s][t][r];
      }
    }
  }
}

// ---- Fused gate GEMM (K=64 -> 256 gates) + LSTM cell update ----------------
// Rows are (i,b) pairs r=i*64+b (64000 valid). AH buffers [i][4096]==[(i,b)][64].
template <bool TWO, bool XPATH>
__global__ void gates_lstm(const _Float16* __restrict__ AH1, const _Float16* __restrict__ W1f,
                           const _Float16* __restrict__ AH2, const _Float16* __restrict__ W2f,
                           const float* __restrict__ AX,  const float* __restrict__ Wx0,
                           const float* __restrict__ b1,  const float* __restrict__ b2,
                           float* __restrict__ Cst, _Float16* __restrict__ HsOut) {
  const int lane = threadIdx.x & 31;
  const int wave = threadIdx.x >> 5;
  const int rt   = blockIdx.x * 8 + wave;       // 16-row tile of (i,b) rows
  const int row  = rt * 16 + (lane & 15);

  v8f acc[16];
#pragma unroll
  for (int t = 0; t < 16; t++)
#pragma unroll
    for (int e = 0; e < 8; e++) acc[t][e] = 0.f;

#pragma unroll
  for (int k0 = 0; k0 < 64; k0 += 32) {
    v16h a1 = load_afrag(AH1, 64, row, k0, lane);
#pragma unroll
    for (int t = 0; t < 16; t++) {
      v16h b = load_bfrag(W1f, 64, t * 16 + (lane & 15), k0, lane);
      acc[t] = __builtin_amdgcn_wmma_f32_16x16x32_f16(
          false, a1, false, b, (short)0, acc[t], false, false);
    }
    if constexpr (TWO) {
      v16h a2 = load_afrag(AH2, 64, row, k0, lane);
#pragma unroll
      for (int t = 0; t < 16; t++) {
        v16h b = load_bfrag(W2f, 64, t * 16 + (lane & 15), k0, lane);
        acc[t] = __builtin_amdgcn_wmma_f32_16x16x32_f16(
            false, a2, false, b, (short)0, acc[t], false, false);
      }
    }
  }

  // Epilogue: gates i/f/o/g for hidden index hp live in the SAME lane
  // (tiles ct, ct+4, ct+8, ct+12) -> fully lane-local LSTM update.
  const int l = lane & 15;
  const int mbase = rt * 16 + ((lane >> 4) & 1) * 8;
#pragma unroll
  for (int r = 0; r < 8; r++) {
    const int m = mbase + r;
    const int inode = m >> 6;
    const int bidx  = m & 63;
    float ax0 = 0.f, ax1 = 0.f;
    if constexpr (XPATH) {
      ax0 = AX[(size_t)inode * 128 + bidx * 2];
      ax1 = AX[(size_t)inode * 128 + bidx * 2 + 1];
    }
#pragma unroll
    for (int ct = 0; ct < 4; ct++) {
      const int hp = ct * 16 + l;
      float iv = acc[ct][r]      + b1[hp]       + b2[hp];
      float fv = acc[ct + 4][r]  + b1[64 + hp]  + b2[64 + hp];
      float ov = acc[ct + 8][r]  + b1[128 + hp] + b2[128 + hp];
      float gv = acc[ct + 12][r] + b1[192 + hp] + b2[192 + hp];
      if constexpr (XPATH) {
        iv += ax0 * Wx0[hp * 2]         + ax1 * Wx0[hp * 2 + 1];
        fv += ax0 * Wx0[(64 + hp) * 2]  + ax1 * Wx0[(64 + hp) * 2 + 1];
        ov += ax0 * Wx0[(128 + hp) * 2] + ax1 * Wx0[(128 + hp) * 2 + 1];
        gv += ax0 * Wx0[(192 + hp) * 2] + ax1 * Wx0[(192 + hp) * 2 + 1];
      }
      const float si = 1.f / (1.f + __expf(-iv));
      const float sf = 1.f / (1.f + __expf(-fv));
      const float so = 1.f / (1.f + __expf(-ov));
      const float tg = tanhf(gv);
      const float co = Cst[(size_t)m * 64 + hp];
      const float cn = sf * co + si * tg;
      const float hn = so * tanhf(cn);
      Cst[(size_t)m * 64 + hp] = cn;
      // node-major transposed state: Hs[b*64+h][node]
      HsOut[(size_t)(bidx * 64 + hp) * NP + inode] = (_Float16)hn;
    }
  }
}

// ---- out[b, k, i] = sum_h h1[b,i,h] * Wp[k,h] + bp[k] ----------------------
__global__ void proj_out(const _Float16* __restrict__ Hs1, const float* __restrict__ Wp,
                         const float* __restrict__ bp, float* __restrict__ out) {
  const int idx = blockIdx.x * 256 + threadIdx.x;    // b*NN + i
  if (idx >= BB * NN) return;
  const int b = idx / NN, i = idx - b * NN;
  float acc[HOR];
#pragma unroll
  for (int k = 0; k < HOR; k++) acc[k] = bp[k];
  for (int h = 0; h < HH; h++) {
    const float hv = (float)Hs1[(size_t)(b * 64 + h) * NP + i];
#pragma unroll
    for (int k = 0; k < HOR; k++) acc[k] += hv * Wp[k * HH + h];
  }
#pragma unroll
  for (int k = 0; k < HOR; k++) out[((size_t)b * HOR + k) * NN + i] = acc[k];
}

extern "C" void kernel_launch(void* const* d_in, const int* in_sizes, int n_in,
                              void* d_out, int out_size, void* d_ws, size_t ws_size,
                              hipStream_t stream) {
  const float* x   = (const float*)d_in[0];
  const float* E1  = (const float*)d_in[1];
  const float* E2  = (const float*)d_in[2];
  const float* Wx0 = (const float*)d_in[3];
  const float* bx0 = (const float*)d_in[4];
  const float* Wh0 = (const float*)d_in[5];
  const float* bh0 = (const float*)d_in[6];
  const float* Wx1 = (const float*)d_in[7];
  const float* bx1 = (const float*)d_in[8];
  const float* Wh1 = (const float*)d_in[9];
  const float* bh1 = (const float*)d_in[10];
  const float* Wp  = (const float*)d_in[11];
  const float* bp  = (const float*)d_in[12];

  char* ws = (char*)d_ws;
  size_t off = 0;
  auto alloc = [&](size_t bytes) {
    void* p = ws + off;
    off += (bytes + 255) & ~(size_t)255;
    return p;
  };
  _Float16* Af   = (_Float16*)alloc((size_t)NP * NP * 2);          // 2 MB
  _Float16* Xs   = (_Float16*)alloc((size_t)TT * 128 * NP * 2);    // 12.6 MB
  _Float16* Hs0  = (_Float16*)alloc((size_t)4096 * NP * 2);        // 8.4 MB
  _Float16* Hs1  = (_Float16*)alloc((size_t)4096 * NP * 2);        // 8.4 MB
  float*    C0   = (float*)   alloc((size_t)64000 * 64 * 4);       // 16.4 MB
  float*    C1   = (float*)   alloc((size_t)64000 * 64 * 4);       // 16.4 MB
  _Float16* AHa  = (_Float16*)alloc((size_t)NP * 4096 * 2);        // 8.4 MB
  _Float16* AHb  = (_Float16*)alloc((size_t)NP * 4096 * 2);
  _Float16* AHc  = (_Float16*)alloc((size_t)NP * 4096 * 2);
  float*    AX   = (float*)   alloc((size_t)NP * 128 * 4);
  _Float16* Wh0f = (_Float16*)alloc((size_t)GG * HH * 2);
  _Float16* Wx1f = (_Float16*)alloc((size_t)GG * HH * 2);
  _Float16* Wh1f = (_Float16*)alloc((size_t)GG * HH * 2);

  // state must be re-initialized every launch (determinism across replays)
  hipMemsetAsync(Hs0, 0, (size_t)4096 * NP * 2, stream);
  hipMemsetAsync(Hs1, 0, (size_t)4096 * NP * 2, stream);
  hipMemsetAsync(C0,  0, (size_t)64000 * 64 * 4, stream);
  hipMemsetAsync(C1,  0, (size_t)64000 * 64 * 4, stream);

  build_A<<<NP, 256, 0, stream>>>(E1, E2, Af);
  build_Xs<<<(TT * 128 * NP) / 256, 256, 0, stream>>>(x, Xs);
  f32_to_f16<<<(GG * HH + 255) / 256, 256, 0, stream>>>(Wh0, Wh0f, GG * HH);
  f32_to_f16<<<(GG * HH + 255) / 256, 256, 0, stream>>>(Wx1, Wx1f, GG * HH);
  f32_to_f16<<<(GG * HH + 255) / 256, 256, 0, stream>>>(Wh1, Wh1f, GG * HH);

  const dim3 blk(256);
  for (int t = 0; t < TT; t++) {
    // layer-0 diffusions: AHa = A @ h0 ; AX = A @ x_t
    wmma_gemm_k1024<true ><<<dim3(4, 64), blk, 0, stream>>>(Af, Hs0, AHa, 4096);
    wmma_gemm_k1024<false><<<dim3(4, 2),  blk, 0, stream>>>(Af, Xs + (size_t)t * 128 * NP, AX, 128);
    // layer-0 gates + cell update -> h0, c0
    gates_lstm<false, true><<<500, blk, 0, stream>>>(AHa, Wh0f, nullptr, nullptr,
                                                     AX, Wx0, bx0, bh0, C0, Hs0);
    // layer-1 diffusions: AHb = A @ h0_new ; AHc = A @ h1
    wmma_gemm_k1024<true><<<dim3(4, 64), blk, 0, stream>>>(Af, Hs0, AHb, 4096);
    wmma_gemm_k1024<true><<<dim3(4, 64), blk, 0, stream>>>(Af, Hs1, AHc, 4096);
    // layer-1 gates (two GEMM paths) + cell update -> h1, c1
    gates_lstm<true, false><<<500, blk, 0, stream>>>(AHb, Wx1f, AHc, Wh1f,
                                                     nullptr, nullptr, bx1, bh1, C1, Hs1);
  }
  proj_out<<<(BB * NN + 255) / 256, 256, 0, stream>>>(Hs1, Wp, bp, (float*)d_out);
}
